// VonMisesSweetNet_25744033972310
// MI455X (gfx1250) — compile-verified, matrix operand order checked by reading, not run on hardware
//
#include <hip/hip_runtime.h>
#include <hip/hip_bf16.h>

typedef __attribute__((ext_vector_type(16))) _Float16 v16h;
typedef __attribute__((ext_vector_type(8)))  float    v8f;
typedef unsigned int u32;
typedef __attribute__((ext_vector_type(4))) u32 u32x4;
typedef __attribute__((ext_vector_type(8))) u32 u32x8;

#define HDIM   128
#define BN_EPS 1e-5f
#define LDA    136   // padded LDS stride (halves) for K=128 rows (64 dw + 4 dw pad)
#define LDZ    72    // padded LDS stride for K=64 rows (32 dw + 4 dw pad)
#define LDV    40    // padded LDS stride for K=32 rows (16 dw + 4 dw pad)

// ---------------------------------------------------------------------------
// WMMA helpers (CDNA5 gfx1250, wave32)
// ---------------------------------------------------------------------------
__device__ __forceinline__ v8f wmma_f16(v16h a, v16h b, v8f c) {
  // (neg_a, A, neg_b, B, c_mod, C, reuse_a, reuse_b)
  return __builtin_amdgcn_wmma_f32_16x16x32_f16(false, a, false, b, (short)0, c,
                                                false, false);
}

// 16-bit A/B fragment loader for 16x16x32 WMMA.
// ISA layout: lane L -> tile row/col = L&15 ; koff = (L>>4)*8
//   halves 0..7  = K[kbase+koff .. +7]
//   halves 8..15 = K[kbase+16+koff .. +7]
__device__ __forceinline__ v16h load_frag(const _Float16* rowptr, int kbase, int lane) {
  const int koff = (lane >> 4) << 3;
  const _Float16* p0 = rowptr + kbase + koff;
  const _Float16* p1 = p0 + 16;
  v16h f;
#pragma unroll
  for (int i = 0; i < 8; ++i) f[i] = p0[i];
#pragma unroll
  for (int i = 0; i < 8; ++i) f[8 + i] = p1[i];
  return f;
}

// ---------------------------------------------------------------------------
// Tensor Data Mover: stage a contiguous f16 matrix [rows][rowDw dwords] from
// global into LDS, inserting a 4-dword pad after each row (padded stride).
// D# packing per CDNA5 ISA.
//   rowDw       : row length in DWORDs (tensor_dim0 = tile_dim0 = stride0)
//   rows        : number of rows (tensor_dim1 = tile_dim1)
//   padIntCode  : pad_interval code, pad every 2^(code+1) dwords == rowDw
// pad_amount code 3 -> 4 dwords pad.
// ---------------------------------------------------------------------------
__device__ __forceinline__ void tdm_load_w(const _Float16* gsrc, _Float16* lds_dst,
                                           u32 rowDw, u32 rows, u32 padIntCode) {
  u32 ldsAddr = (u32)(uintptr_t)lds_dst;  // flat LDS addr: low 32 bits = LDS offset
  unsigned long long ga = (unsigned long long)(uintptr_t)gsrc;
  u32x4 g0;
  g0[0] = 1u;                                             // count=1, user D#
  g0[1] = ldsAddr;                                        // lds_addr
  g0[2] = (u32)ga;                                        // global_addr[31:0]
  g0[3] = ((u32)(ga >> 32) & 0x01FFFFFFu) | (2u << 30);   // global_addr[56:32] | type=2
  u32x8 g1;
  g1[0] = (2u << 16)            // data_size = 4B
        | (1u << 20)            // pad_enable
        | (padIntCode << 22)    // pad_interval
        | (3u << 25);           // pad_amount = 4 dwords
  g1[1] = (rowDw & 0xFFFFu) << 16;                        // tensor_dim0[15:0]
  g1[2] = (rowDw >> 16) | ((rows & 0xFFFFu) << 16);       // td0[31:16] | td1[15:0]
  g1[3] = (rows >> 16) | ((rowDw & 0xFFFFu) << 16);       // td1[31:16] | tile_dim0
  g1[4] = rows & 0xFFFFu;                                 // tile_dim1 (tile_dim2=0)
  g1[5] = rowDw;                                          // tensor_dim0_stride[31:0]
  g1[6] = 0u;
  g1[7] = 0u;
  asm volatile("tensor_load_to_lds %0, %1" :: "s"(g0), "s"(g1) : "memory");
}

// Load 128 rows x cols f32 from global, convert to f16 into padded LDS.
__device__ __forceinline__ void load_rows_f16(_Float16* s, int ld, const float* g,
                                              int rowBase, int n, int cols) {
  const int f4 = cols >> 2;
  for (int i = threadIdx.x; i < 128 * f4; i += (int)blockDim.x) {
    int r = i / f4, c = i - r * f4;
    float4 v = make_float4(0.f, 0.f, 0.f, 0.f);
    int grow = rowBase + r;
    if (grow < n) v = ((const float4*)(g + (size_t)grow * cols))[c];
    _Float16* p = s + r * ld + (c << 2);
    p[0] = (_Float16)v.x; p[1] = (_Float16)v.y;
    p[2] = (_Float16)v.z; p[3] = (_Float16)v.w;
  }
}

// ---------------------------------------------------------------------------
// Embedding gather: h[node] = emb[x[node]]
// ---------------------------------------------------------------------------
__global__ __launch_bounds__(256) void embed_kernel(const int* __restrict__ x,
                                                    const float* __restrict__ emb,
                                                    float* __restrict__ h, int n) {
  int idx = blockIdx.x * blockDim.x + threadIdx.x;
  if (idx >= n * 32) return;
  int node = idx >> 5;
  int c4 = idx & 31;
  int t = x[node];
  ((float4*)(h + (size_t)node * HDIM))[c4] =
      ((const float4*)(emb + (size_t)t * HDIM))[c4];
}

// agg = h  (float4 copy)
__global__ __launch_bounds__(256) void copy4_kernel(const float* __restrict__ src,
                                                    float* __restrict__ dst, int n4) {
  int i = blockIdx.x * blockDim.x + threadIdx.x;
  if (i < n4) ((float4*)dst)[i] = ((const float4*)src)[i];
}

// agg[dst[e]] += h[src[e]]   (thread = one edge x 4 channels)
__global__ __launch_bounds__(256) void scatter_kernel(const float* __restrict__ h,
                                                      const int* __restrict__ srcIdx,
                                                      const int* __restrict__ dstIdx,
                                                      float* __restrict__ agg, int E) {
  int idx = blockIdx.x * blockDim.x + threadIdx.x;
  if (idx >= E * 32) return;
  int e = idx >> 5;
  int c = (idx & 31) << 2;
  int s = srcIdx[e], d = dstIdx[e];
  const float* hp = h + (size_t)s * HDIM + c;
  float* ap = agg + (size_t)d * HDIM + c;
  float4 v = *(const float4*)hp;
  atomicAdd(ap + 0, v.x);
  atomicAdd(ap + 1, v.y);
  atomicAdd(ap + 2, v.z);
  atomicAdd(ap + 3, v.w);
}

// Convert f32 weight [K][Nsrc] (row-major) -> f16 col-major [Npad][K], zero-padded.
__global__ __launch_bounds__(256) void prep_w_kernel(const float* __restrict__ W,
                                                     _Float16* __restrict__ Wt,
                                                     int K, int Nsrc, int Npad) {
  int i = blockIdx.x * blockDim.x + threadIdx.x;
  if (i >= Npad * K) return;
  int nn = i / K, kk = i - nn * K;
  Wt[i] = (nn < Nsrc) ? (_Float16)W[(size_t)kk * Nsrc + nn] : (_Float16)0.f;
}

// ---------------------------------------------------------------------------
// Fused GIN MLP:  out = BN2( BN1(relu(agg@W1+b1)) @ W2 + b2 )
// Block = 256 thr = 8 waves, 128 rows/block; per-wave 16-row strip.
// Weights staged via Tensor Data Mover (padded-stride LDS tiles).
// ---------------------------------------------------------------------------
__global__ __launch_bounds__(256) void gin_mlp_kernel(
    const float* __restrict__ agg,
    const _Float16* __restrict__ W1t, const float* __restrict__ b1,
    const float* __restrict__ bn1g, const float* __restrict__ bn1b,
    const float* __restrict__ bn1m, const float* __restrict__ bn1v,
    const _Float16* __restrict__ W2t, const float* __restrict__ b2,
    const float* __restrict__ bn2g, const float* __restrict__ bn2b,
    const float* __restrict__ bn2m, const float* __restrict__ bn2v,
    float* __restrict__ out, int n) {
  __shared__ _Float16 sW1[HDIM * LDA];
  __shared__ _Float16 sW2[HDIM * LDA];
  __shared__ _Float16 sA[HDIM * LDA];

  const int wave = threadIdx.x >> 5;
  const int lane = threadIdx.x & 31;
  const int rowBase = blockIdx.x * 128;

  if (threadIdx.x == 0) {
    tdm_load_w(W1t, sW1, 64, 128, 5);
    tdm_load_w(W2t, sW2, 64, 128, 5);
  }
  load_rows_f16(sA, LDA, agg, rowBase, n, HDIM);
  if (threadIdx.x == 0) __builtin_amdgcn_s_wait_tensorcnt(0);
  __syncthreads();

  const int m0 = wave * 16;
  const int colLane = lane & 15;
  const int rOff = (lane >> 4) << 3;
  const _Float16* aRow = sA + (m0 + colLane) * LDA;
  const v8f zerov = {};

  // ---- GEMM1: t = relu(A@W1 + b1); t = BN1(t) -> sA (wave-local rows) ----
  v8f acc[8];
#pragma unroll
  for (int nt = 0; nt < 8; ++nt) acc[nt] = zerov;
#pragma unroll
  for (int kt = 0; kt < 4; ++kt) {
    v16h a = load_frag(aRow, kt * 32, lane);
#pragma unroll
    for (int nt = 0; nt < 8; ++nt) {
      v16h b = load_frag(sW1 + (nt * 16 + colLane) * LDA, kt * 32, lane);
      acc[nt] = wmma_f16(a, b, acc[nt]);
    }
  }
#pragma unroll
  for (int nt = 0; nt < 8; ++nt) {
    int col = nt * 16 + colLane;
    float bias = b1[col];
    float sc = bn1g[col] * rsqrtf(bn1v[col] + BN_EPS);
    float sh = bn1b[col] - bn1m[col] * sc;
#pragma unroll
    for (int r = 0; r < 8; ++r) {
      int row = m0 + r + rOff;
      float val = fmaxf(acc[nt][r] + bias, 0.f);
      val = val * sc + sh;
      sA[row * LDA + col] = (_Float16)val;
    }
  }

  // ---- GEMM2: h = BN2(t@W2 + b2) -> global ----
#pragma unroll
  for (int nt = 0; nt < 8; ++nt) acc[nt] = zerov;
#pragma unroll
  for (int kt = 0; kt < 4; ++kt) {
    v16h a = load_frag(aRow, kt * 32, lane);
#pragma unroll
    for (int nt = 0; nt < 8; ++nt) {
      v16h b = load_frag(sW2 + (nt * 16 + colLane) * LDA, kt * 32, lane);
      acc[nt] = wmma_f16(a, b, acc[nt]);
    }
  }
#pragma unroll
  for (int nt = 0; nt < 8; ++nt) {
    int col = nt * 16 + colLane;
    float bias = b2[col];
    float sc = bn2g[col] * rsqrtf(bn2v[col] + BN_EPS);
    float sh = bn2b[col] - bn2m[col] * sc;
#pragma unroll
    for (int r = 0; r < 8; ++r) {
      int row = m0 + r + rOff;
      int grow = rowBase + row;
      float val = (acc[nt][r] + bias) * sc + sh;
      if (grow < n) out[(size_t)grow * HDIM + col] = val;
    }
  }
}

// ---------------------------------------------------------------------------
// Fused heads: von Mises (128->64->64->{10,10,10}) + values (128->32->2)
// ---------------------------------------------------------------------------
__global__ __launch_bounds__(256) void head_kernel(
    const float* __restrict__ h,
    const _Float16* __restrict__ Wv1t, const float* __restrict__ hv_b1,
    const float* __restrict__ hg1, const float* __restrict__ hbe1,
    const float* __restrict__ hm1, const float* __restrict__ hvv1,
    const _Float16* __restrict__ Wv2t, const float* __restrict__ hv_b2,
    const float* __restrict__ hg2, const float* __restrict__ hbe2,
    const float* __restrict__ hm2, const float* __restrict__ hvv2,
    const _Float16* __restrict__ Fct,
    const float* __restrict__ fcw_b, const float* __restrict__ fcm_b,
    const float* __restrict__ fck_b,
    const _Float16* __restrict__ Wval1t, const float* __restrict__ val_b1,
    const float* __restrict__ vg1, const float* __restrict__ vbe1,
    const float* __restrict__ vm1, const float* __restrict__ vv1,
    const _Float16* __restrict__ Wval2t, const float* __restrict__ val_b2,
    const float* __restrict__ vg2, const float* __restrict__ vbe2,
    const float* __restrict__ vm2, const float* __restrict__ vv2,
    float* __restrict__ out, int n) {
  __shared__ _Float16 sH[128 * LDA];
  __shared__ _Float16 sZ[128 * LDZ];
  __shared__ _Float16 sV[128 * LDV];
  __shared__ _Float16 sWv1[64 * LDA];
  __shared__ _Float16 sWv2[64 * LDZ];
  __shared__ _Float16 sFc[48 * LDZ];
  __shared__ _Float16 sWval1[32 * LDA];
  __shared__ _Float16 sWval2[16 * LDV];

  const int wave = threadIdx.x >> 5;
  const int lane = threadIdx.x & 31;
  const int rowBase = blockIdx.x * 128;

  if (threadIdx.x == 0) {
    tdm_load_w(Wv1t,   sWv1,   64, 64, 5);
    tdm_load_w(Wv2t,   sWv2,   32, 64, 4);
    tdm_load_w(Fct,    sFc,    32, 48, 4);
    tdm_load_w(Wval1t, sWval1, 64, 32, 5);
    tdm_load_w(Wval2t, sWval2, 16, 16, 3);
  }
  load_rows_f16(sH, LDA, h, rowBase, n, HDIM);
  if (threadIdx.x == 0) __builtin_amdgcn_s_wait_tensorcnt(0);
  __syncthreads();

  const int m0 = wave * 16;
  const int colLane = lane & 15;
  const int rOff = (lane >> 4) << 3;
  const _Float16* hRow = sH + (m0 + colLane) * LDA;
  const _Float16* zRow = sZ + (m0 + colLane) * LDZ;
  const v8f zerov = {};
  const size_t N10 = (size_t)n * 10;

  // ---- z1 = relu(BN1(h@hv_W1 + b1)) -> sZ ----
  v8f acc4[4];
#pragma unroll
  for (int nt = 0; nt < 4; ++nt) acc4[nt] = zerov;
#pragma unroll
  for (int kt = 0; kt < 4; ++kt) {
    v16h a = load_frag(hRow, kt * 32, lane);
#pragma unroll
    for (int nt = 0; nt < 4; ++nt) {
      v16h b = load_frag(sWv1 + (nt * 16 + colLane) * LDA, kt * 32, lane);
      acc4[nt] = wmma_f16(a, b, acc4[nt]);
    }
  }
#pragma unroll
  for (int nt = 0; nt < 4; ++nt) {
    int col = nt * 16 + colLane;
    float bias = hv_b1[col];
    float sc = hg1[col] * rsqrtf(hvv1[col] + BN_EPS);
    float sh = hbe1[col] - hm1[col] * sc;
#pragma unroll
    for (int r = 0; r < 8; ++r) {
      int row = m0 + r + rOff;
      float val = fmaxf((acc4[nt][r] + bias) * sc + sh, 0.f);
      sZ[row * LDZ + col] = (_Float16)val;
    }
  }

  // ---- z = BN2(z1@hv_W2 + b2) -> sZ (wave-local overwrite) ----
#pragma unroll
  for (int nt = 0; nt < 4; ++nt) acc4[nt] = zerov;
#pragma unroll
  for (int kt = 0; kt < 2; ++kt) {
    v16h a = load_frag(zRow, kt * 32, lane);
#pragma unroll
    for (int nt = 0; nt < 4; ++nt) {
      v16h b = load_frag(sWv2 + (nt * 16 + colLane) * LDZ, kt * 32, lane);
      acc4[nt] = wmma_f16(a, b, acc4[nt]);
    }
  }
#pragma unroll
  for (int nt = 0; nt < 4; ++nt) {
    int col = nt * 16 + colLane;
    float bias = hv_b2[col];
    float sc = hg2[col] * rsqrtf(hvv2[col] + BN_EPS);
    float sh = hbe2[col] - hm2[col] * sc;
#pragma unroll
    for (int r = 0; r < 8; ++r) {
      int row = m0 + r + rOff;
      float val = (acc4[nt][r] + bias) * sc + sh;
      sZ[row * LDZ + col] = (_Float16)val;
    }
  }

  // ---- three fc heads: z @ W[64x10] (padded to 16 cols) ----
  for (int head = 0; head < 3; ++head) {
    v8f acch = zerov;
#pragma unroll
    for (int kt = 0; kt < 2; ++kt) {
      v16h a = load_frag(zRow, kt * 32, lane);
      v16h b = load_frag(sFc + (head * 16 + colLane) * LDZ, kt * 32, lane);
      acch = wmma_f16(a, b, acch);
    }
    int col = colLane;
    if (col < 10) {
      const float* fb = (head == 0) ? fcw_b : (head == 1) ? fcm_b : fck_b;
      float bias = fb[col];
#pragma unroll
      for (int r = 0; r < 8; ++r) {
        int grow = rowBase + m0 + r + rOff;
        float val = acch[r] + bias;
        if (head == 1) val = tanhf(val) * 180.f;
        if (head == 2)
          val = (val > 0.f) ? val + log1pf(__expf(-val)) : log1pf(__expf(val));
        if (grow < n) out[(size_t)head * N10 + (size_t)grow * 10 + col] = val;
      }
    }
  }

  // ---- v1 = relu(BN1(h@val_W1 + b1)) -> sV ----
  v8f accv[2];
#pragma unroll
  for (int nt = 0; nt < 2; ++nt) accv[nt] = zerov;
#pragma unroll
  for (int kt = 0; kt < 4; ++kt) {
    v16h a = load_frag(hRow, kt * 32, lane);
#pragma unroll
    for (int nt = 0; nt < 2; ++nt) {
      v16h b = load_frag(sWval1 + (nt * 16 + colLane) * LDA, kt * 32, lane);
      accv[nt] = wmma_f16(a, b, accv[nt]);
    }
  }
#pragma unroll
  for (int nt = 0; nt < 2; ++nt) {
    int col = nt * 16 + colLane;
    float bias = val_b1[col];
    float sc = vg1[col] * rsqrtf(vv1[col] + BN_EPS);
    float sh = vbe1[col] - vm1[col] * sc;
#pragma unroll
    for (int r = 0; r < 8; ++r) {
      int row = m0 + r + rOff;
      float val = fmaxf((accv[nt][r] + bias) * sc + sh, 0.f);
      sV[row * LDV + col] = (_Float16)val;
    }
  }

  // ---- vals = BN2(v1@val_W2 + b2); sasa = vals[:,0]*2 ; flex = vals[:,1]*2 ----
  {
    v8f accf = zerov;
    v16h a = load_frag(sV + (m0 + colLane) * LDV, 0, lane);
    v16h b = load_frag(sWval2 + colLane * LDV, 0, lane);
    accf = wmma_f16(a, b, accf);
    int col = colLane;
    if (col < 2) {
      float bias = val_b2[col];
      float sc = vg2[col] * rsqrtf(vv2[col] + BN_EPS);
      float sh = vbe2[col] - vm2[col] * sc;
#pragma unroll
      for (int r = 0; r < 8; ++r) {
        int grow = rowBase + m0 + r + rOff;
        float val = ((accf[r] + bias) * sc + sh) * 2.f;
        if (grow < n) out[(size_t)30 * n + (size_t)col * n + grow] = val;
      }
    }
  }
}

// ---------------------------------------------------------------------------
// Launch
// ---------------------------------------------------------------------------
extern "C" void kernel_launch(void* const* d_in, const int* in_sizes, int n_in,
                              void* d_out, int out_size, void* d_ws, size_t ws_size,
                              hipStream_t stream) {
  const int* x = (const int*)d_in[0];
  const int* edge = (const int*)d_in[1];
  const float* emb = (const float*)d_in[2];
  const float* gin_W1 = (const float*)d_in[3];
  const float* gin_b1 = (const float*)d_in[4];
  const float* gin_bn1_g = (const float*)d_in[5];
  const float* gin_bn1_b = (const float*)d_in[6];
  const float* gin_bn1_m = (const float*)d_in[7];
  const float* gin_bn1_v = (const float*)d_in[8];
  const float* gin_W2 = (const float*)d_in[9];
  const float* gin_b2 = (const float*)d_in[10];
  const float* gin_bn2_g = (const float*)d_in[11];
  const float* gin_bn2_b = (const float*)d_in[12];
  const float* gin_bn2_m = (const float*)d_in[13];
  const float* gin_bn2_v = (const float*)d_in[14];
  const float* hv_W1 = (const float*)d_in[15];
  const float* hv_b1 = (const float*)d_in[16];
  const float* hv_bn1_g = (const float*)d_in[17];
  const float* hv_bn1_b = (const float*)d_in[18];
  const float* hv_bn1_m = (const float*)d_in[19];
  const float* hv_bn1_v = (const float*)d_in[20];
  const float* hv_W2 = (const float*)d_in[21];
  const float* hv_b2 = (const float*)d_in[22];
  const float* hv_bn2_g = (const float*)d_in[23];
  const float* hv_bn2_b = (const float*)d_in[24];
  const float* hv_bn2_m = (const float*)d_in[25];
  const float* hv_bn2_v = (const float*)d_in[26];
  const float* fcw_W = (const float*)d_in[27];
  const float* fcw_b = (const float*)d_in[28];
  const float* fcm_W = (const float*)d_in[29];
  const float* fcm_b = (const float*)d_in[30];
  const float* fck_W = (const float*)d_in[31];
  const float* fck_b = (const float*)d_in[32];
  const float* val_W1 = (const float*)d_in[33];
  const float* val_b1 = (const float*)d_in[34];
  const float* val_bn1_g = (const float*)d_in[35];
  const float* val_bn1_b = (const float*)d_in[36];
  const float* val_bn1_m = (const float*)d_in[37];
  const float* val_bn1_v = (const float*)d_in[38];
  const float* val_W2 = (const float*)d_in[39];
  const float* val_b2 = (const float*)d_in[40];
  const float* val_bn2_g = (const float*)d_in[41];
  const float* val_bn2_b = (const float*)d_in[42];
  const float* val_bn2_m = (const float*)d_in[43];
  const float* val_bn2_v = (const float*)d_in[44];

  const int n = in_sizes[0];
  const int E = in_sizes[1] / 2;
  const int* srcIdx = edge;
  const int* dstIdx = edge + E;

  // workspace layout
  float* h = (float*)d_ws;
  float* agg = h + (size_t)n * HDIM;
  _Float16* wp = (_Float16*)(agg + (size_t)n * HDIM);
  _Float16* W1t = wp;                               // 3 * 128*128
  _Float16* W2t = W1t + 3 * HDIM * HDIM;            // 3 * 128*128
  _Float16* Hv1t = W2t + 3 * HDIM * HDIM;           // 64*128
  _Float16* Hv2t = Hv1t + 64 * 128;                 // 64*64
  _Float16* Fct = Hv2t + 64 * 64;                   // 3 * 16*64
  _Float16* Val1t = Fct + 3 * 16 * 64;              // 32*128
  _Float16* Val2t = Val1t + 32 * 128;               // 16*32

  auto prep = [&](const float* W, _Float16* Wt, int K, int Ns, int Np) {
    int tot = Np * K;
    prep_w_kernel<<<(tot + 255) / 256, 256, 0, stream>>>(W, Wt, K, Ns, Np);
  };
  for (int l = 0; l < 3; ++l) {
    prep(gin_W1 + (size_t)l * HDIM * HDIM, W1t + (size_t)l * HDIM * HDIM, 128, 128, 128);
    prep(gin_W2 + (size_t)l * HDIM * HDIM, W2t + (size_t)l * HDIM * HDIM, 128, 128, 128);
  }
  prep(hv_W1, Hv1t, 128, 64, 64);
  prep(hv_W2, Hv2t, 64, 64, 64);
  prep(fcw_W, Fct + 0 * 16 * 64, 64, 10, 16);
  prep(fcm_W, Fct + 1 * 16 * 64, 64, 10, 16);
  prep(fck_W, Fct + 2 * 16 * 64, 64, 10, 16);
  prep(val_W1, Val1t, 128, 32, 32);
  prep(val_W2, Val2t, 32, 2, 16);

  embed_kernel<<<(n * 32 + 255) / 256, 256, 0, stream>>>(x, emb, h, n);

  const int nBlocks = (n + 127) / 128;
  const int n4 = n * (HDIM / 4);
  for (int l = 0; l < 3; ++l) {
    copy4_kernel<<<(n4 + 255) / 256, 256, 0, stream>>>(h, agg, n4);
    scatter_kernel<<<(E * 32 + 255) / 256, 256, 0, stream>>>(h, srcIdx, dstIdx, agg, E);
    gin_mlp_kernel<<<nBlocks, 256, 0, stream>>>(
        agg,
        W1t + (size_t)l * HDIM * HDIM, gin_b1 + l * HDIM,
        gin_bn1_g + l * HDIM, gin_bn1_b + l * HDIM, gin_bn1_m + l * HDIM, gin_bn1_v + l * HDIM,
        W2t + (size_t)l * HDIM * HDIM, gin_b2 + l * HDIM,
        gin_bn2_g + l * HDIM, gin_bn2_b + l * HDIM, gin_bn2_m + l * HDIM, gin_bn2_v + l * HDIM,
        h, n);
  }

  head_kernel<<<nBlocks, 256, 0, stream>>>(
      h,
      Hv1t, hv_b1, hv_bn1_g, hv_bn1_b, hv_bn1_m, hv_bn1_v,
      Hv2t, hv_b2, hv_bn2_g, hv_bn2_b, hv_bn2_m, hv_bn2_v,
      Fct, fcw_b, fcm_b, fck_b,
      Val1t, val_b1, val_bn1_g, val_bn1_b, val_bn1_m, val_bn1_v,
      Val2t, val_b2, val_bn2_g, val_bn2_b, val_bn2_m, val_bn2_v,
      (float*)d_out, n);
}
